// LuongAttention_75720273428849
// MI455X (gfx1250) — matrix operand order repaired and checked by reading, not compiled, not run on hardware
//
#include <hip/hip_runtime.h>

typedef _Float16 h16 __attribute__((ext_vector_type(16)));
typedef _Float16 h4  __attribute__((ext_vector_type(4)));
typedef float    f8  __attribute__((ext_vector_type(8)));
typedef float    f4  __attribute__((ext_vector_type(4)));
typedef unsigned int u32x4 __attribute__((ext_vector_type(4)));
typedef int          i32x4 __attribute__((ext_vector_type(4)));
typedef int          i32x8 __attribute__((ext_vector_type(8)));

#define B_      32
#define S_      4096
#define H_      1024
#define CHUNK   64
#define NCH     (S_ / CHUNK)     // 64 chunks per batch
#define THREADS 256

// ---------------------------------------------------------------------------
// Kernel 1: per (batch, s-chunk) partial attention.
// The 64x1024 f32 tile of values[b] is streamed HBM -> LDS by the Tensor Data
// Mover (one descriptor posted by wave 0, tracked by TENSORcnt), then scores
// are computed with v_wmma_f32_16x16x32_f16 (f32->f16 conversion at operand
// build; f32 accumulate), followed by local softmax stats and the
// un-normalized partial context — all from the single LDS-resident tile.
// values is read from HBM exactly once: ~512 MB -> ~22 us floor at 23.3 TB/s.
// ---------------------------------------------------------------------------
__global__ void __launch_bounds__(THREADS)
luong_attn_partial(const float* __restrict__ query,
                   const float* __restrict__ values,
                   const float* __restrict__ scale,
                   float* __restrict__ wsStats,   // [B*NCH][2] = (m, l)
                   float* __restrict__ wsCtx)     // [B*NCH][H] partial context
{
    extern __shared__ char smemRaw[];
    float*    vlsF   = (float*)smemRaw;              // CHUNK * H_ f32 (256 KB)
    _Float16* qh     = (_Float16*)(vlsF + CHUNK * H_); // H_ halves
    float*    sc     = (float*)(qh + H_);            // CHUNK scores
    float*    pr     = sc + CHUNK;                   // CHUNK probs
    float*    sscale = pr + CHUNK;                   // CHUNK scale values

    const int b   = blockIdx.x;
    const int ch  = blockIdx.y;
    const int s0  = ch * CHUNK;
    const int tid = threadIdx.x;

    const float* vb = values + ((size_t)b * S_ + s0) * H_;

    // ---- stage values tile HBM -> LDS via the Tensor Data Mover ----
#if defined(__gfx1250__) && __has_builtin(__builtin_amdgcn_tensor_load_to_lds)
    if (tid < 32) {   // one wave posts the DMA descriptor
        u32x4 g0;
        const unsigned long long ga = (unsigned long long)(uintptr_t)vb;
        g0.x = 1u;                                    // count=1 valid descriptor
        g0.y = (unsigned)(uintptr_t)(void*)vlsF;      // lds_addr (LDS byte offset)
        g0.z = (unsigned)(ga & 0xFFFFFFFFu);          // global_addr[31:0]
        g0.w = (unsigned)((ga >> 32) & 0x01FFFFFFu)   // global_addr[56:32]
             | (2u << 30);                            // type=2 ("image")

        i32x8 g1;
        g1[0] = 0x20000;             // data_size=2 (4B elems); no multicast/pad
        g1[1] = (int)(H_ << 16);     // tensor_dim0[15:0]=1024 (bits 63:48)
        g1[2] = (int)(CHUNK << 16);  // tensor_dim0 hi=0 | tensor_dim1[15:0]=64
        g1[3] = (int)(H_ << 16);     // tensor_dim1 hi=0 | tile_dim0=1024
        g1[4] = CHUNK;               // tile_dim1=64 | tile_dim2=0
        g1[5] = H_;                  // tensor_dim0_stride[31:0]=1024
        g1[6] = 0;                   // dim0_stride hi | dim1_stride lo
        g1[7] = 0;                   // dim1_stride hi

        i32x4 gz = {0, 0, 0, 0};     // 2D tensor: groups 2/3 unused
#if __clang_major__ >= 23
        i32x8 gz8 = {0, 0, 0, 0, 0, 0, 0, 0};
        __builtin_amdgcn_tensor_load_to_lds(g0, g1, gz, gz, gz8, 0);
#else
        __builtin_amdgcn_tensor_load_to_lds(g0, g1, gz, gz, 0);
#endif
#if __has_builtin(__builtin_amdgcn_s_wait_tensorcnt)
        __builtin_amdgcn_s_wait_tensorcnt(0);
#endif
    }
#else
    // Fallback: manual cooperative staging (f32).
    for (int i = tid; i < CHUNK * H_ / 4; i += THREADS) {
        __builtin_prefetch(vb + (size_t)i * 4 + THREADS * 16, 0, 1);
        *(f4*)(vlsF + (size_t)i * 4) = *(const f4*)(vb + (size_t)i * 4);
    }
#endif

    // ---- stage query[b,:] as f16 (tiny) ----
    const float* qb = query + (size_t)b * H_;
    for (int i = tid; i < H_ / 4; i += THREADS) {
        f4 v = *(const f4*)(qb + i * 4);
        h4 hh = { (_Float16)v.x, (_Float16)v.y, (_Float16)v.z, (_Float16)v.w };
        *(h4*)(qh + i * 4) = hh;
    }
    // ---- stage scale[s0:s0+CHUNK] ----
    if (tid < CHUNK) sscale[tid] = scale[s0 + tid];
    __syncthreads();

    // ---- Phase B: scores via WMMA. Waves 0..3 each own a 16-row s tile ----
    const int wave = tid >> 5;     // wave32
    const int lane = tid & 31;
    if (wave < CHUNK / 16) {
        const int row = (wave << 4) + (lane & 15);   // A: M = lane&15
        const int grp = lane >> 4;                   // lane half selects K group
        f8 acc = {};
        for (int k0 = 0; k0 < H_; k0 += 32) {
            h16 a, bmat;
            const float* ar = vlsF + (size_t)row * H_ + k0;
            const _Float16* qr = qh + k0;
#pragma unroll
            for (int e = 0; e < 16; ++e) {
                const int v  = e >> 1;
                // 16-bit A 16x32 layout (ISA 7.12.2): VGPR v<4 -> K = 8*grp + 2v + (e&1),
                // VGPR v>=4 -> K += 16.
                const int ka = 16 * (v >> 2) + 8 * grp + 2 * (v & 3) + (e & 1);
                a[e] = (_Float16)ar[ka];             // f32 LDS -> f16 operand
                // B 32x16: same K order as A (K = 8*grp + e, +16 for e>=8);
                // query replicated across all 16 columns -> every D column = score.
                bmat[e] = qr[ka];
            }
            acc = __builtin_amdgcn_wmma_f32_16x16x32_f16(
                false, a, false, bmat, (short)0, acc, false, false);
        }
        // D layout: VGPR i, lanes 0-15 -> M=i (N=lane), lanes 16-31 -> M=i+8.
        // Take column N==0 (lanes 0 and 16).
        if ((lane & 15) == 0) {
#pragma unroll
            for (int i = 0; i < 8; ++i) {
                const int srow = (wave << 4) + (grp << 3) + i;
                sc[srow] = acc[i] * sscale[srow];
            }
        }
    }
    __syncthreads();

    // ---- Phase C: local softmax stats (redundant per thread; 64 LDS reads) ----
    float m = -3.4e38f;
    for (int s = 0; s < CHUNK; ++s) m = fmaxf(m, sc[s]);
    float l = 0.0f;
    for (int s = 0; s < CHUNK; ++s) l += __expf(sc[s] - m);
    if (tid < CHUNK) pr[tid] = __expf(sc[tid] - m);
    __syncthreads();

    // ---- Phase D: un-normalized partial context from the f32 LDS tile ----
    float a0 = 0.f, a1 = 0.f, a2 = 0.f, a3 = 0.f;
    for (int s = 0; s < CHUNK; ++s) {
        const float p = pr[s];
        const float* vr = vlsF + (size_t)s * H_;
        a0 += p * vr[tid];
        a1 += p * vr[tid + 256];
        a2 += p * vr[tid + 512];
        a3 += p * vr[tid + 768];
    }
    float* pc = wsCtx + ((size_t)(b * NCH + ch)) * H_;
    pc[tid]       = a0;
    pc[tid + 256] = a1;
    pc[tid + 512] = a2;
    pc[tid + 768] = a3;
    if (tid == 0) {
        wsStats[(b * NCH + ch) * 2 + 0] = m;
        wsStats[(b * NCH + ch) * 2 + 1] = l;
    }
}

// ---------------------------------------------------------------------------
// Kernel 2: softmax merge across the 64 chunks of each batch.
// context[b,h] = sum_i exp(m_i - M) * pc_i[h] / sum_i l_i * exp(m_i - M)
// Traffic: 32 * 64 * 1024 floats = 8.4 MB — negligible.
// ---------------------------------------------------------------------------
__global__ void __launch_bounds__(THREADS)
luong_attn_reduce(const float* __restrict__ wsStats,
                  const float* __restrict__ wsCtx,
                  float* __restrict__ out)
{
    const int b   = blockIdx.x;
    const int tid = threadIdx.x;
    __shared__ float w[NCH];

    float M = -3.4e38f;
    for (int i = 0; i < NCH; ++i) M = fmaxf(M, wsStats[(b * NCH + i) * 2]);
    float L = 0.0f;
    for (int i = 0; i < NCH; ++i)
        L += wsStats[(b * NCH + i) * 2 + 1] * __expf(wsStats[(b * NCH + i) * 2] - M);
    if (tid < NCH) w[tid] = __expf(wsStats[(b * NCH + tid) * 2] - M);
    __syncthreads();

    const float inv = 1.0f / L;
    float a0 = 0.f, a1 = 0.f, a2 = 0.f, a3 = 0.f;
    for (int i = 0; i < NCH; ++i) {
        const float wi = w[i];
        const float* pc = wsCtx + ((size_t)(b * NCH + i)) * H_;
        a0 += wi * pc[tid];
        a1 += wi * pc[tid + 256];
        a2 += wi * pc[tid + 512];
        a3 += wi * pc[tid + 768];
    }
    float* ob = out + (size_t)b * H_;
    ob[tid]       = a0 * inv;
    ob[tid + 256] = a1 * inv;
    ob[tid + 512] = a2 * inv;
    ob[tid + 768] = a3 * inv;
}

extern "C" void kernel_launch(void* const* d_in, const int* in_sizes, int n_in,
                              void* d_out, int out_size, void* d_ws, size_t ws_size,
                              hipStream_t stream) {
    const float* query  = (const float*)d_in[0];   // [B, H]
    const float* values = (const float*)d_in[1];   // [B, S, H]
    const float* scale  = (const float*)d_in[2];   // [S, 1]
    float* out = (float*)d_out;                    // [B, H]

    float* wsStats = (float*)d_ws;                 // B*NCH*2 floats
    float* wsCtx   = wsStats + (size_t)B_ * NCH * 2;  // B*NCH*H floats (~8.4 MB)

    const size_t shmem = (size_t)CHUNK * H_ * sizeof(float)      // values tile (f32)
                       + (size_t)H_ * sizeof(_Float16)           // query (f16)
                       + (size_t)3 * CHUNK * sizeof(float);      // scores/probs/scale

    dim3 grid(B_, NCH);
    luong_attn_partial<<<grid, THREADS, shmem, stream>>>(query, values, scale,
                                                         wsStats, wsCtx);
    luong_attn_reduce<<<B_, THREADS, 0, stream>>>(wsStats, wsCtx, out);
}
